// SpatialClassifierVN_83786222011212
// MI455X (gfx1250) — matrix operand order repaired and checked by reading, not compiled, not run on hardware
//
#include <hip/hip_runtime.h>
#include <hip/hip_bf16.h>

// MI455X (gfx1250) implementation of SpatialClassifierVN.
// All dense contractions run on v_wmma_f32_16x16x32_f16 (f16 in, f32 acc).
// Tile = 16 edges/nodes per 128-thread (4 wave32) workgroup.
// Param leaves are assumed flattened in recursive insertion order of the
// setup_inputs() dict (41 device inputs total).

typedef __attribute__((ext_vector_type(16))) _Float16 v16h;
typedef __attribute__((ext_vector_type(8)))  float    v8f;
typedef _Float16 h16;

#define THREADS   128
#define N_WAVES   4
#define EPS_VN    1e-6f
#define CUTOFF_R  10.0f
#define SLOPE_VN  0.2f
#define SLOPE_SCA 0.01f
#define PI_F      3.14159265358979f

// ---------------- WMMA fragment helpers (ISA 7.12.2 layouts) ----------------

// A fragment: 16x32 f16 tile from row-major LDS (ldx halves per row).
// lane&15 = row M; K = {k0..k0+7} U {k0+16..k0+23}, +8 for lanes 16..31.
__device__ __forceinline__ v16h ldA(const h16* X, int ldx, int m0, int k0) {
  const int lane = threadIdx.x & 31;
  const h16* q = X + (m0 + (lane & 15)) * ldx + (k0 + ((lane >> 4) << 3));
  v16h a;
#pragma unroll
  for (int i = 0; i < 8; ++i) a[i] = q[i];
#pragma unroll
  for (int i = 0; i < 8; ++i) a[8 + i] = q[16 + i];
  return a;
}

// B fragment: 32x16 tile of W^T, W row-major (N rows, K cols) f16 in global.
// B[k][n] = W[n][k]; lane holds col n=lane&15, K = {k0..k0+7} U {k0+16..k0+23}, +8 for lanes 16..31.
__device__ __forceinline__ v16h ldB(const h16* W, int K, int n0, int k0) {
  const int lane = threadIdx.x & 31;
  const h16* q = W + (n0 + (lane & 15)) * K + (k0 + ((lane >> 4) << 3));
  v16h b;
#pragma unroll
  for (int i = 0; i < 8; ++i) b[i] = q[i];
#pragma unroll
  for (int i = 0; i < 8; ++i) b[8 + i] = q[16 + i];
  return b;
}

// C/D: VGPR r -> row m0 + r + 8*(lane>=16), col n0 + (lane&15)
__device__ __forceinline__ void stC(float* O, int ldo, int m0, int n0, v8f c) {
  const int lane = threadIdx.x & 31;
  float* q = O + (m0 + ((lane >> 4) << 3)) * ldo + (n0 + (lane & 15));
#pragma unroll
  for (int r = 0; r < 8; ++r) q[r * ldo] = c[r];
}

// OUT(MxN) = X(MxK) @ W^T (+ bias per output column).  N-tiles split over waves.
// K-outer loop with register-resident accumulators for all of this wave's
// (mt, nt) tiles: A fragments reused across nt, B fragments reused across mt.
// Call sites pass literal M/N/K; forceinline makes all loop bounds constant so
// acc[]/a[] scalarize into VGPRs (no dynamic indexing).  The bias add is
// pointer-guarded (uniform branch) so nullptr call sites emit zero VALU.
__device__ __forceinline__ void gemm(const h16* X, int ldx, const h16* W,
                                     const float* bias, float* O, int ldo,
                                     int M, int N, int K) {
  const int wave = threadIdx.x >> 5;
  if ((N & 63) == 0) {                 // N multiple of 64: uniform tiles/wave
    const int MT  = M >> 4;            // compile-time after inlining (<=3)
    const int NTL = N >> 6;            // compile-time after inlining (<=4)
    v8f acc[12];
#pragma unroll
    for (int i = 0; i < MT * NTL; ++i) acc[i] = (v8f){};
    for (int kt = 0; kt < K; kt += 32) {
      v16h a[3];
#pragma unroll
      for (int mt = 0; mt < MT; ++mt) a[mt] = ldA(X, ldx, mt * 16, kt);
#pragma unroll
      for (int j = 0; j < NTL; ++j) {
        v16h b = ldB(W, K, wave * 16 + j * 64, kt);
#pragma unroll
        for (int mt = 0; mt < MT; ++mt)
          acc[mt * NTL + j] = __builtin_amdgcn_wmma_f32_16x16x32_f16(
              false, a[mt], false, b, (short)0, acc[mt * NTL + j], false, false);
      }
    }
#pragma unroll
    for (int j = 0; j < NTL; ++j) {
#pragma unroll
      for (int mt = 0; mt < MT; ++mt) {
        v8f c = acc[mt * NTL + j];
        if (bias) {
          float bv = bias[wave * 16 + j * 64 + (threadIdx.x & 15)];
#pragma unroll
          for (int r = 0; r < 8; ++r) c[r] += bv;
        }
        stC(O, ldo, mt * 16, wave * 16 + j * 64, c);
      }
    }
  } else {                             // generic path (only the 16x32 output)
    for (int nt = wave * 16; nt < N; nt += N_WAVES * 16) {
      for (int mt = 0; mt < M; mt += 16) {
        v8f c = {};
        for (int kt = 0; kt < K; kt += 32) {
          v16h a = ldA(X, ldx, mt, kt);
          v16h b = ldB(W, K, nt, kt);
          c = __builtin_amdgcn_wmma_f32_16x16x32_f16(false, a, false, b,
                                                     (short)0, c, false, false);
        }
        if (bias) {
          float bv = bias[nt + (threadIdx.x & 15)];
#pragma unroll
          for (int r = 0; r < 8; ++r) c[r] += bv;
        }
        stC(O, ldo, mt, nt, c);
      }
    }
  }
}

__device__ __forceinline__ float sigmoidf_(float x) { return 1.0f / (1.0f + expf(-x)); }

// ---------------- Param bundle ----------------

struct Params {
  const h16 *nWv1, *nWv2, *nWs, *nWg;                      // msg.node_gvl
  const h16 *eWv1, *eWv2, *eWs, *eWg, *eWdir;              // msg.edge_gvp
  const h16 *Wsca, *We2n, *Wn2e, *Wev;                     // msg mixers
  const h16 *oWv1, *oWv2, *oWs, *oWg;                      // msg.out_gvl
  const h16 *cWv1, *cWv2, *cWs, *cWg, *cWdir;              // cls.gvp
  const h16 *gWv1, *gWs;                                   // cls.gvl
  const float *eeW, *bgN, *bgE, *bsca, *be2n, *bn2e, *bgO, *bgC;
  h16 *nsca;      // (Nc,256) f16
  h16 *nvec;      // (Nc,3,64) f16
  float *y0;      // (Nq,256) f32 accumulators
  float *y1;      // (Nq,3,64) f32 accumulators
};

// ---------------- Small utility kernels ----------------

__global__ void k_cast(const float* __restrict__ s, h16* __restrict__ d, int n) {
  int i = blockIdx.x * 256 + threadIdx.x;
  if (i < n) d[i] = (h16)s[i];
}

__global__ void k_zero(float* __restrict__ p, int n) {
  int i = blockIdx.x * 256 + threadIdx.x;
  if (i < n) p[i] = 0.0f;
}

// ---------------- Node precompute: gv_linear(node_gvl) ----------------
// n_sca = concat(|Wv1 v|, sca) @ Ws^T ; n_vec = sigmoid(n_sca@Wg^T+bg) * (Wv2 Wv1 v)

__global__ __launch_bounds__(THREADS)
void k_node(const float* __restrict__ node_sca, const float* __restrict__ node_vec,
            Params P, int Nc) {
  extern __shared__ char sm[];
  h16*   hV  = (h16*)(sm + 0);          // 48x64  (A staging)
  h16*   hX  = (h16*)(sm + 6144);       // 16x320 (concat input)
  h16*   hHs = (h16*)(sm + 16384);      // 16x256 (out_sca f16)
  float* vA  = (float*)(sm + 24576);    // 48x64
  float* vB  = (float*)(sm + 36864);    // 48x64
  float* sT  = (float*)(sm + 49152);    // 16x256
  float* sG  = (float*)(sm + 65536);    // 16x64
  const int tid = threadIdx.x;
  const int n0 = blockIdx.x * 16;

  for (int i = tid; i < 48 * 64; i += THREADS) {
    int row = i >> 6, c = i & 63, n = row / 3, d = row % 3;
    int nn = n0 + n; nn = nn < Nc ? nn : Nc - 1;
    hV[i] = (h16)node_vec[(nn * 64 + c) * 3 + d];
  }
  __syncthreads();
  gemm(hV, 64, P.nWv1, nullptr, vA, 64, 48, 64, 64);      // v_inter
  __syncthreads();
  for (int i = tid; i < 48 * 64; i += THREADS) hV[i] = (h16)vA[i];
  for (int i = tid; i < 16 * 64; i += THREADS) {
    int n = i >> 6, h = i & 63;
    float a = vA[(3 * n) * 64 + h], b = vA[(3 * n + 1) * 64 + h], c = vA[(3 * n + 2) * 64 + h];
    hX[n * 320 + h] = (h16)sqrtf(a * a + b * b + c * c);
  }
  for (int i = tid; i < 16 * 256; i += THREADS) {
    int n = i >> 8, j = i & 255;
    int nn = n0 + n; nn = nn < Nc ? nn : Nc - 1;
    hX[n * 320 + 64 + j] = (h16)node_sca[nn * 256 + j];
  }
  __syncthreads();
  gemm(hX, 320, P.nWs, nullptr, sT, 256, 16, 256, 320);   // out_sca
  __syncthreads();
  for (int i = tid; i < 16 * 256; i += THREADS) hHs[i] = (h16)sT[i];
  __syncthreads();
  gemm(hV, 64, P.nWv2, nullptr, vB, 64, 48, 64, 64);      // out_vec
  gemm(hHs, 256, P.nWg, P.bgN, sG, 64, 16, 64, 256);      // gate (pre-sigmoid)
  __syncthreads();
  for (int i = tid; i < 16 * 64; i += THREADS) sG[i] = sigmoidf_(sG[i]);
  __syncthreads();
  for (int i = tid; i < 16 * 256; i += THREADS) {
    int n = i >> 8, nn = n0 + n;
    if (nn < Nc) P.nsca[nn * 256 + (i & 255)] = hHs[i];
  }
  for (int i = tid; i < 48 * 64; i += THREADS) {
    int row = i >> 6, h = i & 63, n = row / 3, d = row % 3, nn = n0 + n;
    if (nn < Nc) P.nvec[(nn * 3 + d) * 64 + h] = (h16)(sG[n * 64 + h] * vB[i]);
  }
}

// ---------------- Fused edge-message kernel ----------------

__global__ __launch_bounds__(THREADS)
void k_edge(const float* __restrict__ pq, const float* __restrict__ pc,
            const int* __restrict__ eq, const int* __restrict__ ec,
            Params P, int E) {
  extern __shared__ char sm[];
  h16*   hV   = (h16*)(sm + 0);          // 48x64  vector A staging (multi-use)
  h16*   hXe  = (h16*)(sm + 6144);       // 16x128 edge gvl concat input
  h16*   hS1  = (h16*)(sm + 10240);      // 16x64  scalar A staging
  h16*   hNs  = (h16*)(sm + 12288);      // 16x256 gathered n_sca f16
  h16*   hXo  = (h16*)(sm + 20480);      // 16x320 out_gvl concat input
  h16*   hHs  = (h16*)(sm + 30720);      // 16x256 h_sca f16
  float* vA   = (float*)(sm + 38912);    // 48x64
  float* vB   = (float*)(sm + 51200);    // 48x64
  float* CT   = (float*)(sm + 63488);    // 16x256 (also holds 48x64 dmat/w_ev)
  float* sS1  = (float*)(sm + 79872);    // 16x64
  float* sS2  = (float*)(sm + 83968);    // 16x64
  float* sG   = (float*)(sm + 88064);    // 16x64
  float* s_dist = (float*)(sm + 92160);  // 16
  float* s_C    = (float*)(sm + 92224);  // 16
  float* s_vd   = (float*)(sm + 92288);  // 48
  int*   s_eq   = (int*)(sm + 92480);    // 16
  int*   s_ec   = (int*)(sm + 92544);    // 16
  const int tid = threadIdx.x;
  const int e0 = blockIdx.x * 16;

  // --- geometry ---
  if (tid < 16) {
    int ee = e0 + tid;
    bool ok = ee < E;
    int ie = ok ? ee : 0;
    int qi = eq[ie], ci = ec[ie];
    s_eq[tid] = qi; s_ec[tid] = ci;
    float vx = pq[qi * 3 + 0] - pc[ci * 3 + 0];
    float vy = pq[qi * 3 + 1] - pc[ci * 3 + 1];
    float vz = pq[qi * 3 + 2] - pc[ci * 3 + 2];
    if (!ok) { vx = vy = vz = 0.f; }
    float d = sqrtf(vx * vx + vy * vy + vz * vz);
    s_dist[tid] = d;
    float inv = 1.0f / (d + 1e-7f);
    s_vd[tid * 3 + 0] = vx * inv;
    s_vd[tid * 3 + 1] = vy * inv;
    s_vd[tid * 3 + 2] = vz * inv;
    float C = 0.5f * (cosf(d * PI_F / CUTOFF_R) + 1.0f);
    C = (d <= CUTOFF_R) ? C : 0.0f;
    s_C[tid] = ok ? C : 0.0f;
  }
  __syncthreads();

  // --- gaussian smearing -> hXe cols 64..127, edge vec features -> hV ---
  const float gstep = CUTOFF_R / 63.0f;
  const float gcoef = -0.5f / (gstep * gstep);
  for (int i = tid; i < 16 * 64; i += THREADS) {
    int e = i >> 6, j = i & 63;
    float diff = s_dist[e] - gstep * (float)j;
    hXe[e * 128 + 64 + j] = (h16)expf(gcoef * diff * diff);
  }
  for (int i = tid; i < 48 * 64; i += THREADS) {
    int row = i >> 6, c = i & 63, e = row / 3, d = row % 3;
    hV[i] = (h16)(P.eeW[c] * s_vd[e * 3 + d]);
  }
  __syncthreads();

  // --- edge GVP: gv_linear ---
  gemm(hV, 64, P.eWv1, nullptr, vA, 64, 48, 64, 64);        // v_inter
  __syncthreads();
  for (int i = tid; i < 48 * 64; i += THREADS) hV[i] = (h16)vA[i];
  for (int i = tid; i < 16 * 64; i += THREADS) {
    int e = i >> 6, h = i & 63;
    float a = vA[(3 * e) * 64 + h], b = vA[(3 * e + 1) * 64 + h], c = vA[(3 * e + 2) * 64 + h];
    hXe[e * 128 + h] = (h16)sqrtf(a * a + b * b + c * c);    // v_norm
  }
  __syncthreads();
  gemm(hXe, 128, P.eWs, nullptr, sS1, 64, 16, 64, 128);      // out_sca_e
  __syncthreads();
  for (int i = tid; i < 16 * 64; i += THREADS) hS1[i] = (h16)sS1[i];
  __syncthreads();
  gemm(hV, 64, P.eWv2, nullptr, vB, 64, 48, 64, 64);         // out_vec_e
  gemm(hS1, 64, P.eWg, P.bgE, sG, 64, 16, 64, 64);           // gate
  __syncthreads();
  for (int i = tid; i < 16 * 64; i += THREADS) sG[i] = sigmoidf_(sG[i]);
  __syncthreads();
  for (int i = tid; i < 48 * 64; i += THREADS) {             // x = gate * out_vec
    int e = (i >> 6) / 3, h = i & 63;
    float x = sG[e * 64 + h] * vB[i];
    vB[i] = x; hV[i] = (h16)x;
  }
  for (int i = tid; i < 16 * 64; i += THREADS) {             // e_sca = leaky(out_sca)
    float s = sS1[i]; s = s >= 0.f ? s : SLOPE_SCA * s;
    sS1[i] = s; hS1[i] = (h16)s;
  }
  __syncthreads();
  gemm(hV, 64, P.eWdir, nullptr, CT, 64, 48, 64, 64);        // dmat = Wdir x
  __syncthreads();
  for (int i = tid; i < 16 * 64; i += THREADS) {             // VNLeakyReLU -> e_vec (hV)
    int e = i >> 6, h = i & 63;
    float x0 = vB[(3 * e) * 64 + h], x1 = vB[(3 * e + 1) * 64 + h], x2 = vB[(3 * e + 2) * 64 + h];
    float d0 = CT[(3 * e) * 64 + h], d1 = CT[(3 * e + 1) * 64 + h], d2 = CT[(3 * e + 2) * 64 + h];
    float dot = x0 * d0 + x1 * d1 + x2 * d2;
    float dd = d0 * d0 + d1 * d1 + d2 * d2;
    float o0, o1, o2;
    if (dot >= 0.f) { o0 = x0; o1 = x1; o2 = x2; }
    else {
      float t = dot / (dd + EPS_VN);
      o0 = SLOPE_VN * x0 + (1.f - SLOPE_VN) * (x0 - t * d0);
      o1 = SLOPE_VN * x1 + (1.f - SLOPE_VN) * (x1 - t * d1);
      o2 = SLOPE_VN * x2 + (1.f - SLOPE_VN) * (x2 - t * d2);
    }
    hV[(3 * e) * 64 + h] = (h16)o0;
    hV[(3 * e + 1) * 64 + h] = (h16)o1;
    hV[(3 * e + 2) * 64 + h] = (h16)o2;
  }
  for (int i = tid; i < 16 * 256; i += THREADS) {            // gather n_sca
    int e = i >> 8;
    hNs[i] = P.nsca[(size_t)s_ec[e] * 256 + (i & 255)];
  }
  __syncthreads();

  // --- mixing: y_sca ---
  gemm(hS1, 64, P.Wsca, P.bsca, CT, 256, 16, 256, 64);       // t = e_sca@Wsca^T+bsca
  __syncthreads();
  for (int i = tid; i < 16 * 256; i += THREADS) {            // y_sca -> hXo cols 64..
    int e = i >> 8, j = i & 255;
    hXo[e * 320 + 64 + j] = (h16)((float)hNs[i] * CT[i]);
  }
  __syncthreads();
  // --- mixing: y_vec ---
  gemm(hS1, 64, P.We2n, P.be2n, sG, 64, 16, 64, 64);         // g1
  gemm(hNs, 256, P.Wn2e, P.bn2e, sS2, 64, 16, 64, 256);      // g2
  gemm(hV, 64, P.Wev, nullptr, CT, 64, 48, 64, 64);          // w_ev = Wev e_vec
  __syncthreads();
  for (int i = tid; i < 48 * 64; i += THREADS) {             // y_vec -> hV
    int row = i >> 6, h = i & 63, e = row / 3, d = row % 3;
    float nv = (float)P.nvec[((size_t)s_ec[e] * 3 + d) * 64 + h];
    hV[i] = (h16)(sG[e * 64 + h] * nv + sS2[e * 64 + h] * CT[i]);
  }
  __syncthreads();

  // --- out_gvl ---
  gemm(hV, 64, P.oWv1, nullptr, vA, 64, 48, 64, 64);         // v_inter2
  __syncthreads();
  for (int i = tid; i < 48 * 64; i += THREADS) hV[i] = (h16)vA[i];
  for (int i = tid; i < 16 * 64; i += THREADS) {
    int e = i >> 6, h = i & 63;
    float a = vA[(3 * e) * 64 + h], b = vA[(3 * e + 1) * 64 + h], c = vA[(3 * e + 2) * 64 + h];
    hXo[e * 320 + h] = (h16)sqrtf(a * a + b * b + c * c);
  }
  __syncthreads();
  gemm(hXo, 320, P.oWs, nullptr, CT, 256, 16, 256, 320);     // h_sca
  __syncthreads();
  for (int i = tid; i < 16 * 256; i += THREADS) hHs[i] = (h16)CT[i];
  __syncthreads();
  gemm(hV, 64, P.oWv2, nullptr, vB, 64, 48, 64, 64);         // out_vec2
  gemm(hHs, 256, P.oWg, P.bgO, sG, 64, 16, 64, 256);         // gate
  __syncthreads();
  for (int i = tid; i < 16 * 64; i += THREADS) sG[i] = sigmoidf_(sG[i]);
  __syncthreads();

  // --- cutoff scaling + scatter-add ---
  for (int i = tid; i < 16 * 256; i += THREADS) {
    int e = i >> 8;
    float v = CT[i] * s_C[e];
    atomicAdd(&P.y0[(size_t)s_eq[e] * 256 + (i & 255)], v);
  }
  for (int i = tid; i < 48 * 64; i += THREADS) {
    int row = i >> 6, h = i & 63, e = row / 3, d = row % 3;
    float v = sG[e * 64 + h] * vB[i] * s_C[e];
    atomicAdd(&P.y1[((size_t)s_eq[e] * 3 + d) * 64 + h], v);
  }
}

// ---------------- Classifier: gv_perceptron(cls.gvp) + gv_linear(cls.gvl) ----------------

__global__ __launch_bounds__(THREADS)
void k_cls(Params P, float* __restrict__ out, int Nq) {
  extern __shared__ char sm[];
  h16*   hV  = (h16*)(sm + 0);          // 48x64
  h16*   hX  = (h16*)(sm + 6144);       // 16x320
  h16*   hHs = (h16*)(sm + 16384);      // 16x256
  float* vA  = (float*)(sm + 24576);    // 48x64
  float* vB  = (float*)(sm + 36864);    // 48x64
  float* sT  = (float*)(sm + 49152);    // 16x256
  float* sG  = (float*)(sm + 65536);    // 16x64
  float* sO  = (float*)(sm + 69632);    // 16x32
  const int tid = threadIdx.x;
  const int q0 = blockIdx.x * 16;

  for (int i = tid; i < 48 * 64; i += THREADS) {
    int row = i >> 6, q = row / 3;
    int qq = q0 + q; qq = qq < Nq ? qq : Nq - 1;
    hV[i] = (h16)P.y1[((size_t)qq * 3 + (row % 3)) * 64 + (i & 63)];
  }
  __syncthreads();
  gemm(hV, 64, P.cWv1, nullptr, vA, 64, 48, 64, 64);         // v_inter
  __syncthreads();
  for (int i = tid; i < 48 * 64; i += THREADS) hV[i] = (h16)vA[i];
  for (int i = tid; i < 16 * 64; i += THREADS) {
    int q = i >> 6, h = i & 63;
    float a = vA[(3 * q) * 64 + h], b = vA[(3 * q + 1) * 64 + h], c = vA[(3 * q + 2) * 64 + h];
    hX[q * 320 + h] = (h16)sqrtf(a * a + b * b + c * c);
  }
  for (int i = tid; i < 16 * 256; i += THREADS) {
    int q = i >> 8;
    int qq = q0 + q; qq = qq < Nq ? qq : Nq - 1;
    hX[q * 320 + 64 + (i & 255)] = (h16)P.y0[(size_t)qq * 256 + (i & 255)];
  }
  __syncthreads();
  gemm(hX, 320, P.cWs, nullptr, sT, 256, 16, 256, 320);      // s1
  __syncthreads();
  for (int i = tid; i < 16 * 256; i += THREADS) hHs[i] = (h16)sT[i];
  __syncthreads();
  gemm(hV, 64, P.cWv2, nullptr, vB, 64, 48, 64, 64);         // out_vec
  gemm(hHs, 256, P.cWg, P.bgC, sG, 64, 16, 64, 256);         // gate
  __syncthreads();
  for (int i = tid; i < 16 * 64; i += THREADS) sG[i] = sigmoidf_(sG[i]);
  __syncthreads();
  for (int i = tid; i < 48 * 64; i += THREADS) {             // v = gate*out_vec
    int q = (i >> 6) / 3, h = i & 63;
    float x = sG[q * 64 + h] * vB[i];
    vB[i] = x; hV[i] = (h16)x;
  }
  for (int i = tid; i < 16 * 256; i += THREADS) {            // s = leaky(s1)
    float s = sT[i]; sT[i] = s >= 0.f ? s : SLOPE_SCA * s;
  }
  __syncthreads();
  gemm(hV, 64, P.cWdir, nullptr, vA, 64, 48, 64, 64);        // dmat
  __syncthreads();
  for (int i = tid; i < 16 * 64; i += THREADS) {             // VNLeakyReLU -> hV
    int q = i >> 6, h = i & 63;
    float x0 = vB[(3 * q) * 64 + h], x1 = vB[(3 * q + 1) * 64 + h], x2 = vB[(3 * q + 2) * 64 + h];
    float d0 = vA[(3 * q) * 64 + h], d1 = vA[(3 * q + 1) * 64 + h], d2 = vA[(3 * q + 2) * 64 + h];
    float dot = x0 * d0 + x1 * d1 + x2 * d2;
    float dd = d0 * d0 + d1 * d1 + d2 * d2;
    float o0, o1, o2;
    if (dot >= 0.f) { o0 = x0; o1 = x1; o2 = x2; }
    else {
      float t = dot / (dd + EPS_VN);
      o0 = SLOPE_VN * x0 + (1.f - SLOPE_VN) * (x0 - t * d0);
      o1 = SLOPE_VN * x1 + (1.f - SLOPE_VN) * (x1 - t * d1);
      o2 = SLOPE_VN * x2 + (1.f - SLOPE_VN) * (x2 - t * d2);
    }
    hV[(3 * q) * 64 + h] = (h16)o0;
    hV[(3 * q + 1) * 64 + h] = (h16)o1;
    hV[(3 * q + 2) * 64 + h] = (h16)o2;
  }
  __syncthreads();
  gemm(hV, 64, P.gWv1, nullptr, vB, 64, 48, 64, 64);         // v_inter3
  __syncthreads();
  for (int i = tid; i < 16 * 64; i += THREADS) {
    int q = i >> 6, h = i & 63;
    float a = vB[(3 * q) * 64 + h], b = vB[(3 * q + 1) * 64 + h], c = vB[(3 * q + 2) * 64 + h];
    hX[q * 320 + h] = (h16)sqrtf(a * a + b * b + c * c);
  }
  for (int i = tid; i < 16 * 256; i += THREADS)
    hX[(i >> 8) * 320 + 64 + (i & 255)] = (h16)sT[i];
  __syncthreads();
  gemm(hX, 320, P.gWs, nullptr, sO, 32, 16, 32, 320);        // y_cls
  __syncthreads();
  for (int i = tid; i < 16 * 32; i += THREADS) {
    int q = i >> 5, qq = q0 + q;
    if (qq < Nq) out[qq * 32 + (i & 31)] = sO[i];
  }
}

// ---------------- Host launcher ----------------

extern "C" void kernel_launch(void* const* d_in, const int* in_sizes, int n_in,
                              void* d_out, int out_size, void* d_ws, size_t ws_size,
                              hipStream_t stream) {
  const float* pos_q    = (const float*)d_in[0];
  const float* pos_c    = (const float*)d_in[1];
  const float* node_sca = (const float*)d_in[2];
  const float* node_vec = (const float*)d_in[3];
  const int*   edge_q   = (const int*)d_in[4];
  const int*   edge_c   = (const int*)d_in[5];
  const int Nq = in_sizes[0] / 3;
  const int Nc = in_sizes[1] / 3;
  const int E  = in_sizes[4];

  // f16 weight list: {d_in index, N(out rows), K(in cols)}
  struct WSpec { int idx, n, k; };
  const WSpec wl[24] = {
    {7, 64, 64},  {8, 64, 64},  {9, 256, 320}, {10, 64, 256},       // node_gvl
    {12, 64, 64}, {13, 64, 64}, {14, 64, 128}, {15, 64, 64}, {17, 64, 64}, // edge_gvp
    {18, 256, 64}, {20, 64, 64}, {22, 64, 256}, {24, 64, 64},       // mixers
    {25, 64, 64}, {26, 64, 64}, {27, 256, 320}, {28, 64, 256},      // out_gvl
    {30, 64, 64}, {31, 64, 64}, {32, 256, 320}, {33, 64, 256}, {35, 64, 64}, // cls.gvp
    {36, 64, 64}, {38, 32, 320}                                     // cls.gvl
  };

  char* ws = (char*)d_ws;
  size_t off = 0;
  auto carve = [&](size_t bytes) -> char* {
    char* p = ws + off;
    off = (off + bytes + 255) & ~(size_t)255;
    return p;
  };

  h16* wdst[24];
  for (int i = 0; i < 24; ++i)
    wdst[i] = (h16*)carve((size_t)wl[i].n * wl[i].k * sizeof(h16));
  h16*   nsca = (h16*)carve((size_t)Nc * 256 * sizeof(h16));
  h16*   nvec = (h16*)carve((size_t)Nc * 3 * 64 * sizeof(h16));
  float* y0   = (float*)carve((size_t)Nq * 256 * sizeof(float));
  float* y1   = (float*)carve((size_t)Nq * 3 * 64 * sizeof(float));

  for (int i = 0; i < 24; ++i) {
    int n = wl[i].n * wl[i].k;
    k_cast<<<(n + 255) / 256, 256, 0, stream>>>((const float*)d_in[wl[i].idx], wdst[i], n);
  }
  {
    int n0f = Nq * 256, n1f = Nq * 3 * 64;
    k_zero<<<(n0f + 255) / 256, 256, 0, stream>>>(y0, n0f);
    k_zero<<<(n1f + 255) / 256, 256, 0, stream>>>(y1, n1f);
  }

  Params P;
  P.nWv1 = wdst[0];  P.nWv2 = wdst[1];  P.nWs = wdst[2];   P.nWg = wdst[3];
  P.eWv1 = wdst[4];  P.eWv2 = wdst[5];  P.eWs = wdst[6];   P.eWg = wdst[7];  P.eWdir = wdst[8];
  P.Wsca = wdst[9];  P.We2n = wdst[10]; P.Wn2e = wdst[11]; P.Wev = wdst[12];
  P.oWv1 = wdst[13]; P.oWv2 = wdst[14]; P.oWs = wdst[15];  P.oWg = wdst[16];
  P.cWv1 = wdst[17]; P.cWv2 = wdst[18]; P.cWs = wdst[19];  P.cWg = wdst[20]; P.cWdir = wdst[21];
  P.gWv1 = wdst[22]; P.gWs  = wdst[23];
  P.eeW  = (const float*)d_in[6];
  P.bgN  = (const float*)d_in[11];
  P.bgE  = (const float*)d_in[16];
  P.bsca = (const float*)d_in[19];
  P.be2n = (const float*)d_in[21];
  P.bn2e = (const float*)d_in[23];
  P.bgO  = (const float*)d_in[29];
  P.bgC  = (const float*)d_in[34];
  P.nsca = nsca; P.nvec = nvec; P.y0 = y0; P.y1 = y1;

  const size_t SM_NODE = 69632;
  const size_t SM_EDGE = 92672;
  const size_t SM_CLS  = 71680;

  k_node<<<(Nc + 15) / 16, THREADS, SM_NODE, stream>>>(node_sca, node_vec, P, Nc);
  k_edge<<<(E + 15) / 16, THREADS, SM_EDGE, stream>>>(pos_q, pos_c, edge_q, edge_c, P, E);
  k_cls <<<(Nq + 15) / 16, THREADS, SM_CLS, stream>>>(P, (float*)d_out, Nq);
}
